// DecoderLayer_52252572123883
// MI455X (gfx1250) — matrix operand order repaired and checked by reading, not compiled, not run on hardware
//
#include <hip/hip_runtime.h>
#include <hip/hip_bf16.h>
#include <math.h>

// ---------------------------------------------------------------------------
// Transformer decoder layer for MI455X (gfx1250, wave32, WMMA, async DMA).
//
// GEMM operands are pre-converted to f16 in global memory (halves streamed
// bytes; fp32 accumulate via WMMA keeps accuracy).  GEMM blocks (256 thr /
// 8 waves, 128x64 macro-tile) stage A/W k-slabs into double-buffered LDS
// with global_load_async_to_lds_b128: each per-lane 16B packet is one WMMA
// fragment k-group, so the swizzle into fragment order is pure addressing —
// the DMA engine does the layout.  Inner loop: issue next slab (3 async ops)
// -> s_wait_asynccnt 3 -> barrier -> 8x ds_load_b128 -> 4x wmma -> barrier.
//
// Fragment layouts (CDNA5 ISA 7.12.2, wave32):
//   A 16x32 f16 : k-group g of row r -> lane = r%16 + (g&1)*16, h0 = (g>=2)*8
//   B 32x16 f16 : k-group g of col n -> lane = n%16 + (g&1)*16, h0 = (g>=2)*8
//   C 16x16 f32 : vgpr r, lane l -> (M = r + (l<16?0:8), N = l%16)
// ---------------------------------------------------------------------------

typedef __attribute__((ext_vector_type(16))) _Float16 v16h;
typedef __attribute__((ext_vector_type(8)))  _Float16 v8h;
typedef __attribute__((ext_vector_type(8)))  float    v8f;

__device__ __forceinline__ int a_frag_k(int lane, int h) {
    return h + ((lane < 16) ? 0 : 16);
}
__device__ __forceinline__ int b_frag_k(int lane, int h) {
    return h + ((lane < 16) ? 0 : 16);
}

// per-lane async 16B copy global -> LDS (tracked by ASYNCcnt)
__device__ __forceinline__ void async_b128(const _Float16* lds_dst,
                                           const _Float16* gsrc) {
    unsigned loff = (unsigned)(uintptr_t)lds_dst;          // low 32b = LDS addr
    unsigned long long ga = (unsigned long long)(uintptr_t)gsrc;
    asm volatile("global_load_async_to_lds_b128 %0, %1, off"
                 :: "v"(loff), "v"(ga) : "memory");
}

// ---------------------------------------------------------------------------
// f32 -> f16 elementwise (n multiple of 2048; 8 elements per thread)
// ---------------------------------------------------------------------------
__global__ void __launch_bounds__(256)
f32_to_f16_kernel(const float* __restrict__ s, _Float16* __restrict__ d) {
    const size_t i = ((size_t)blockIdx.x * 256 + threadIdx.x) * 8;
    const float4* s4 = (const float4*)(s + i);
    const float4 f0 = s4[0], f1 = s4[1];
    v8h p;
    p[0]=(_Float16)f0.x; p[1]=(_Float16)f0.y; p[2]=(_Float16)f0.z; p[3]=(_Float16)f0.w;
    p[4]=(_Float16)f1.x; p[5]=(_Float16)f1.y; p[6]=(_Float16)f1.z; p[7]=(_Float16)f1.w;
    *(v8h*)(d + i) = p;
}

// ---------------------------------------------------------------------------
// C[M,N] = A[M,K](f16) @ W[N,K](f16)^T + bias (+ residual) (opt relu)
// Outputs: Cf (f32, optional) and/or Ch (f16, optional).
// ---------------------------------------------------------------------------
__global__ void __launch_bounds__(256)
gemm_wmma_kernel(const _Float16* __restrict__ A,
                 const _Float16* __restrict__ W,
                 const float* __restrict__ bias,
                 const float* __restrict__ residual,
                 float* __restrict__ Cf,
                 _Float16* __restrict__ Ch,
                 int M, int N, int K, int do_relu) {
    __shared__ _Float16 ldsA[2][8 * 32 * 16];   // 2 x 8 KB, fragment order
    __shared__ _Float16 ldsB[2][4 * 32 * 16];   // 2 x 4 KB, fragment order

    const int t    = threadIdx.x;
    const int lane = t & 31;
    const int wave = t >> 5;
    const int wm   = wave & 3;
    const int wn   = wave >> 2;
    const int blockM = blockIdx.y * 128;
    const int blockN = blockIdx.x * 64;

    // --- async staging roles (per-lane packets of 8 halves = 16 B) ---
    // A slab 128x32: packets p = 2t, 2t+1 -> row rA = t>>1, groups gA0, gA0+1
    const int rA  = t >> 1;
    const int gA0 = (t & 1) * 2;
    const _Float16* srcA = A + (size_t)(blockM + rA) * K + gA0 * 8;
    const int mtA = rA >> 4, rrA = rA & 15;
    const int offA0 = (mtA * 32 + rrA + ((gA0     & 1) ? 16 : 0)) * 16 + ((gA0     >= 2) ? 8 : 0);
    const int offA1 = (mtA * 32 + rrA + (((gA0+1) & 1) ? 16 : 0)) * 16 + (((gA0+1) >= 2) ? 8 : 0);
    // B slab 64x32: packet p = t -> row nB = t>>2, group gB = t&3
    const int nB = t >> 2, gB = t & 3;
    const _Float16* srcB = W + (size_t)(blockN + nB) * K + gB * 8;
    const int offB = ((nB >> 4) * 32 + (nB & 15) + ((gB & 1) ? 16 : 0)) * 16
                   + ((gB >= 2) ? 8 : 0);

    v8f acc00 = {}, acc01 = {}, acc10 = {}, acc11 = {};

    // prologue: stage slab 0 into buffer 0
    async_b128(&ldsA[0][offA0], srcA);
    async_b128(&ldsA[0][offA1], srcA + 8);
    async_b128(&ldsB[0][offB],  srcB);

    int buf = 0;
    for (int k0 = 0; k0 < K; k0 += 32, buf ^= 1) {
        if (k0 + 32 < K) {               // overlap next slab DMA with compute
            async_b128(&ldsA[buf ^ 1][offA0], srcA + k0 + 32);
            async_b128(&ldsA[buf ^ 1][offA1], srcA + k0 + 40);
            async_b128(&ldsB[buf ^ 1][offB],  srcB + k0 + 32);
            asm volatile("s_wait_asynccnt 0x3" ::: "memory");  // current slab done
        } else {
            asm volatile("s_wait_asynccnt 0x0" ::: "memory");
        }
        __syncthreads();                 // all waves' DMA for this slab done

        const v16h a0 = *(const v16h*)&ldsA[buf][((wm * 2 + 0) * 32 + lane) * 16];
        const v16h a1 = *(const v16h*)&ldsA[buf][((wm * 2 + 1) * 32 + lane) * 16];
        const v16h b0 = *(const v16h*)&ldsB[buf][((wn * 2 + 0) * 32 + lane) * 16];
        const v16h b1 = *(const v16h*)&ldsB[buf][((wn * 2 + 1) * 32 + lane) * 16];
        acc00 = __builtin_amdgcn_wmma_f32_16x16x32_f16(false, a0, false, b0, (short)0, acc00, false, false);
        acc01 = __builtin_amdgcn_wmma_f32_16x16x32_f16(false, a0, false, b1, (short)0, acc01, false, false);
        acc10 = __builtin_amdgcn_wmma_f32_16x16x32_f16(false, a1, false, b0, (short)0, acc10, false, false);
        acc11 = __builtin_amdgcn_wmma_f32_16x16x32_f16(false, a1, false, b1, (short)0, acc11, false, false);
        __syncthreads();                 // done reading buf before it refills
    }

    // --- epilogue: 2x2 tiles ---
    const int mofs = (lane < 16) ? 0 : 8;
    const int ncol = lane & 15;
#pragma unroll
    for (int i = 0; i < 2; ++i) {
#pragma unroll
        for (int j = 0; j < 2; ++j) {
            const v8f acc = (i == 0) ? (j == 0 ? acc00 : acc01)
                                     : (j == 0 ? acc10 : acc11);
            const int m0 = blockM + (wm * 2 + i) * 16;
            const int n  = blockN + (wn * 2 + j) * 16 + ncol;
            const float bn = bias ? bias[n] : 0.0f;
#pragma unroll
            for (int r = 0; r < 8; ++r) {
                const int m = m0 + r + mofs;
                float v = acc[r] + bn;
                if (residual) v += residual[(size_t)m * N + n];
                if (do_relu)  v = v > 0.0f ? v : 0.0f;
                if (Cf) Cf[(size_t)m * N + n] = v;
                if (Ch) Ch[(size_t)m * N + n] = (_Float16)v;
            }
        }
    }
}

// ---------------------------------------------------------------------------
// scores[b,h,q,k] = (Q . K) * scale, masked with -1e9.  Q/K f32 [B,S,H*DK].
// ---------------------------------------------------------------------------
#define WPB 4
__global__ void attn_scores_kernel(const float* __restrict__ Q,
                                   const float* __restrict__ Km,
                                   const unsigned char* __restrict__ mask, // [B,Sq,Sk]
                                   float* __restrict__ scores,             // [B,H,Sq,Sk]
                                   int Sq, int Sk, int H, int DM, int DK,
                                   float scale) {
    const int lane = threadIdx.x & 31;
    const int wave = threadIdx.x >> 5;
    const int bh = blockIdx.z;
    const int b  = bh / H, h = bh % H;
    const int q0 = (blockIdx.y * WPB + wave) * 16;
    const int k0v = blockIdx.x * 16;

    const int row = lane & 15;
    const float* Qrow = Q  + ((size_t)b * Sq + (q0 + row)) * DM + h * DK;
    const float* Krow = Km + ((size_t)b * Sk + (k0v + row)) * DM + h * DK;

    v8f acc = {};
    for (int k0 = 0; k0 < DK; k0 += 32) {
        v16h a, bfr;
#pragma unroll
        for (int hh = 0; hh < 16; ++hh) {
            a[hh]   = (_Float16)Qrow[k0 + a_frag_k(lane, hh)];
            bfr[hh] = (_Float16)Krow[k0 + b_frag_k(lane, hh)];
        }
        acc = __builtin_amdgcn_wmma_f32_16x16x32_f16(false, a, false, bfr,
                                                     (short)0, acc, false, false);
    }

    const int mofs = (lane < 16) ? 0 : 8;
    const int n = k0v + (lane & 15);
    float* out = scores + ((size_t)bh * Sq) * Sk;
#pragma unroll
    for (int r = 0; r < 8; ++r) {
        const int m = q0 + r + mofs;
        float v = acc[r] * scale;
        if (mask[((size_t)b * Sq + m) * Sk + n]) v = -1e9f;
        out[(size_t)m * Sk + n] = v;
    }
}

// ---------------------------------------------------------------------------
// ctx[b,q,h,d] = sum_k probs[b,h,q,k] * V[b,k,h,d]; ctx emitted as f16
// (it is only ever consumed as a GEMM A operand).  V 32x16 k-tile staged in
// LDS in B-fragment order, shared by the 4 waves of the block.
// ---------------------------------------------------------------------------
__global__ void __launch_bounds__(128)
attn_ctx_kernel(const float* __restrict__ P,      // [B,H,Sq,Sk]
                const float* __restrict__ V,      // [B,Sk,H*DV] f32
                _Float16* __restrict__ ctx,       // [B,Sq,H*DV] f16
                int Sq, int Sk, int H, int DM, int DV) {
    __shared__ _Float16 ldsV[32 * 16];

    const int t    = threadIdx.x;
    const int lane = t & 31;
    const int wave = t >> 5;
    const int bh = blockIdx.z;
    const int b  = bh / H, h = bh % H;
    const int q0 = (blockIdx.y * WPB + wave) * 16;
    const int n0 = blockIdx.x * 16;

    const int row = lane & 15;
    const float* Prow = P + ((size_t)bh * Sq + (q0 + row)) * Sk;
    const int ncolbase = h * DV + n0;
    const int ncol = ncolbase + (lane & 15);

    const int sn  = t & 15;
    const int skk = t >> 4;

    v8f acc = {};
    for (int k0 = 0; k0 < Sk; k0 += 32) {
        __syncthreads();
#pragma unroll
        for (int i = 0; i < 4; ++i) {
            const int kk = skk + 8 * i;
            const int dl = sn + ((kk >= 16) ? 16 : 0);
            ldsV[dl * 16 + (kk & 15)] =
                (_Float16)V[((size_t)b * Sk + k0 + kk) * DM + ncolbase + sn];
        }
        __syncthreads();

        v16h a;
#pragma unroll
        for (int hh = 0; hh < 16; ++hh)
            a[hh] = (_Float16)Prow[k0 + a_frag_k(lane, hh)];
        const v16h bfr = *(const v16h*)&ldsV[lane * 16];
        acc = __builtin_amdgcn_wmma_f32_16x16x32_f16(false, a, false, bfr,
                                                     (short)0, acc, false, false);
    }

    const int mofs = (lane < 16) ? 0 : 8;
#pragma unroll
    for (int r = 0; r < 8; ++r) {
        const int m = q0 + r + mofs;
        ctx[((size_t)b * Sq + m) * DM + ncol] = (_Float16)acc[r];
    }
}

// ---------------------------------------------------------------------------
// In-place row softmax: data[rows, cols]
// ---------------------------------------------------------------------------
__global__ void softmax_rows_kernel(float* __restrict__ data, int cols) {
    float* p = data + (size_t)blockIdx.x * cols;
    __shared__ float red[256];
    const int t = threadIdx.x;

    float mx = -INFINITY;
    for (int c = t; c < cols; c += 256) mx = fmaxf(mx, p[c]);
    red[t] = mx; __syncthreads();
    for (int s = 128; s > 0; s >>= 1) {
        if (t < s) red[t] = fmaxf(red[t], red[t + s]);
        __syncthreads();
    }
    mx = red[0]; __syncthreads();

    float sum = 0.0f;
    for (int c = t; c < cols; c += 256) {
        float e = __expf(p[c] - mx);
        p[c] = e;
        sum += e;
    }
    red[t] = sum; __syncthreads();
    for (int s = 128; s > 0; s >>= 1) {
        if (t < s) red[t] += red[t + s];
        __syncthreads();
    }
    const float inv = 1.0f / red[0];
    for (int c = t; c < cols; c += 256) p[c] *= inv;
}

// ---------------------------------------------------------------------------
// Row LayerNorm: Y = (X - mu)*rsqrt(var+eps)*g + b ; optional f16 shadow Yh.
// ---------------------------------------------------------------------------
__global__ void layernorm_kernel(const float* __restrict__ X,
                                 const float* __restrict__ g,
                                 const float* __restrict__ bb,
                                 float* __restrict__ Y,
                                 _Float16* __restrict__ Yh, int D) {
    const float* x = X + (size_t)blockIdx.x * D;
    float* y = Y + (size_t)blockIdx.x * D;
    __shared__ float rs[256], rs2[256];
    const int t = threadIdx.x;

    float s = 0.0f, s2 = 0.0f;
    for (int c = t; c < D; c += 256) {
        float v = x[c];
        s += v; s2 += v * v;
    }
    rs[t] = s; rs2[t] = s2; __syncthreads();
    for (int st = 128; st > 0; st >>= 1) {
        if (t < st) { rs[t] += rs[t + st]; rs2[t] += rs2[t + st]; }
        __syncthreads();
    }
    const float mu  = rs[0] / D;
    const float var = rs2[0] / D - mu * mu;
    const float inv = rsqrtf(var + 1e-5f);
    for (int c = t; c < D; c += 256) {
        const float v = (x[c] - mu) * inv * g[c] + bb[c];
        y[c] = v;
        if (Yh) Yh[(size_t)blockIdx.x * D + c] = (_Float16)v;
    }
}

// ---------------------------------------------------------------------------
extern "C" void kernel_launch(void* const* d_in, const int* in_sizes, int n_in,
                              void* d_out, int out_size, void* d_ws, size_t ws_size,
                              hipStream_t stream) {
    (void)in_sizes; (void)n_in; (void)out_size; (void)ws_size;

    const int Bb = 4, S = 1024, DM = 1024, H = 16, DK = 64, DV = 64, DFF = 4096;
    const int M = Bb * S;
    const float scale = 1.0f / 8.0f;

    const float* dec_in   = (const float*)d_in[0];
    const float* enc_out  = (const float*)d_in[1];
    const unsigned char* mask_self = (const unsigned char*)d_in[2];
    const unsigned char* mask_enc  = (const unsigned char*)d_in[3];
    const float* Wq1 = (const float*)d_in[4];  const float* bq1 = (const float*)d_in[5];
    const float* Wk1 = (const float*)d_in[6];  const float* bk1 = (const float*)d_in[7];
    const float* Wv1 = (const float*)d_in[8];  const float* bv1 = (const float*)d_in[9];
    const float* Wo1 = (const float*)d_in[10]; const float* bo1 = (const float*)d_in[11];
    const float* g1  = (const float*)d_in[12]; const float* be1 = (const float*)d_in[13];
    const float* Wq2 = (const float*)d_in[14]; const float* bq2 = (const float*)d_in[15];
    const float* Wk2 = (const float*)d_in[16]; const float* bk2 = (const float*)d_in[17];
    const float* Wv2 = (const float*)d_in[18]; const float* bv2 = (const float*)d_in[19];
    const float* Wo2 = (const float*)d_in[20]; const float* bo2 = (const float*)d_in[21];
    const float* g2  = (const float*)d_in[22]; const float* be2 = (const float*)d_in[23];
    const float* W1  = (const float*)d_in[24]; const float* b1  = (const float*)d_in[25];
    const float* W2  = (const float*)d_in[26]; const float* b2  = (const float*)d_in[27];
    const float* g3  = (const float*)d_in[28]; const float* be3 = (const float*)d_in[29];

    // ---- workspace ----
    // f32 slabs: qb kb vb tmp x1 x2  (6 * MD floats)
    // f16 region: hx (x0h/x1h/x2h) | ench | ctxh | weights (4*MD halves)
    // hid f16 (4*MD halves) reuses the qb+kb f32 slabs during the FFN.
    float* ws = (float*)d_ws;
    const size_t MD = (size_t)M * DM;             // 4,194,304
    const size_t QW = (size_t)DM * DM;            // 1,048,576
    float* qb  = ws + 0 * MD;
    float* kb  = ws + 1 * MD;
    float* vb  = ws + 2 * MD;
    float* tmp = ws + 3 * MD;
    float* x1  = ws + 4 * MD;
    float* x2  = ws + 5 * MD;
    _Float16* hb   = (_Float16*)(ws + 6 * MD);
    _Float16* hx   = hb;                          // f16 of current query stream
    _Float16* ench = hb + MD;
    _Float16* hctx = hb + 2 * MD;
    _Float16* hw   = hb + 3 * MD;                 // f16 weights, 4*MD halves
    _Float16* Wq1h = hw + 0 * QW; _Float16* Wk1h = hw + 1 * QW;
    _Float16* Wv1h = hw + 2 * QW; _Float16* Wo1h = hw + 3 * QW;
    _Float16* Wq2h = hw + 4 * QW; _Float16* Wk2h = hw + 5 * QW;
    _Float16* Wv2h = hw + 6 * QW; _Float16* Wo2h = hw + 7 * QW;
    _Float16* W1h  = hw + 8 * QW;                 // DFF*DM = 4*QW
    _Float16* W2h  = hw + 12 * QW;                // DM*DFF = 4*QW
    _Float16* hidh = (_Float16*)ws;               // [M,DFF] f16 over qb..kb

    float* xout  = (float*)d_out;
    float* attn1 = xout + MD;
    float* attn2 = attn1 + (size_t)Bb * H * S * S;

    const dim3 blk128(128);
    const dim3 blk256(256);

    auto conv = [&](const float* s, _Float16* d, size_t n) {
        f32_to_f16_kernel<<<dim3((unsigned)(n / 2048)), blk256, 0, stream>>>(s, d);
    };
    auto gemm = [&](const _Float16* A, const _Float16* W, const float* bias,
                    const float* res, float* Cf, _Float16* Ch,
                    int m, int n, int k, int relu) {
        dim3 grid(n / 64, m / 128);
        gemm_wmma_kernel<<<grid, blk256, 0, stream>>>(A, W, bias, res, Cf, Ch,
                                                      m, n, k, relu);
    };

    // ---- one-time f16 conversions ----
    conv(dec_in, hx, MD);
    conv(enc_out, ench, MD);
    conv(Wq1, Wq1h, QW); conv(Wk1, Wk1h, QW); conv(Wv1, Wv1h, QW); conv(Wo1, Wo1h, QW);
    conv(Wq2, Wq2h, QW); conv(Wk2, Wk2h, QW); conv(Wv2, Wv2h, QW); conv(Wo2, Wo2h, QW);
    conv(W1, W1h, 4 * QW); conv(W2, W2h, 4 * QW);

    auto mha = [&](const _Float16* xqh, const _Float16* xkvh,
                   const float* res_f32, const unsigned char* mask,
                   const _Float16* Wqh, const float* bq,
                   const _Float16* Wkh, const float* bk,
                   const _Float16* Wvh, const float* bv,
                   const _Float16* Woh, const float* bo,
                   const float* lg, const float* lb,
                   float* probs, float* xnext, _Float16* xnexth) {
        gemm(xqh,  Wqh, bq, nullptr, qb, nullptr, M, DM, DM, 0);
        gemm(xkvh, Wkh, bk, nullptr, kb, nullptr, M, DM, DM, 0);
        gemm(xkvh, Wvh, bv, nullptr, vb, nullptr, M, DM, DM, 0);
        {
            dim3 grid(S / 16, S / (16 * WPB), Bb * H);
            attn_scores_kernel<<<grid, blk128, 0, stream>>>(qb, kb, mask, probs,
                                                            S, S, H, DM, DK, scale);
        }
        softmax_rows_kernel<<<dim3(Bb * H * S), blk256, 0, stream>>>(probs, S);
        {
            dim3 grid(DV / 16, S / (16 * WPB), Bb * H);
            attn_ctx_kernel<<<grid, blk128, 0, stream>>>(probs, vb, hctx,
                                                         S, S, H, DM, DV);
        }
        gemm(hctx, Woh, bo, res_f32, tmp, nullptr, M, DM, DM, 0);
        layernorm_kernel<<<dim3(M), blk256, 0, stream>>>(tmp, lg, lb,
                                                         xnext, xnexth, DM);
    };

    // 1) masked self-attention: dec_in -> x1 (hx becomes f16(x1)); probs -> attn1
    mha(hx, hx, dec_in, mask_self, Wq1h, bq1, Wk1h, bk1, Wv1h, bv1, Wo1h, bo1,
        g1, be1, attn1, x1, hx);
    // 2) enc-dec attention: x1 queries, enc K/V -> x2 (hx becomes f16(x2))
    mha(hx, ench, x1, mask_enc, Wq2h, bq2, Wk2h, bk2, Wv2h, bv2, Wo2h, bo2,
        g2, be2, attn2, x2, hx);
    // 3) FFN: hid = relu(x2 @ W1^T + b1) (f16); tmp = hid @ W2^T + b2 + x2; LN
    gemm(hx, W1h, b1, nullptr, nullptr, hidh, M, DFF, DM, 1);
    gemm(hidh, W2h, b2, x2, tmp, nullptr, M, DM, DFF, 0);
    layernorm_kernel<<<dim3(M), blk256, 0, stream>>>(tmp, g3, be3, xout, nullptr, DM);
}
